// MLP_43250320670752
// MI455X (gfx1250) — matrix operand order repaired and checked by reading, not compiled, not run on hardware
//
#include <hip/hip_runtime.h>

// ---------------------------------------------------------------------------
// MLP: out[512, 256000] = relu(relu(embed_sum + b1) @ W2 + b2) @ W3 + b3
// Dominant cost: final GEMM (M=512, N=256000, K=256), 67.1 GFLOP,
// 262 MB W3 read + 524 MB out write  ->  ~34 us HBM floor @ 23.3 TB/s.
// Needs ~2 PFLOP/s effective matrix rate -> f16 WMMA (v_wmma_f32_16x16x32_f16)
// with fp32 accumulation. W3 is read exactly once (each block covers full M).
// ---------------------------------------------------------------------------

typedef __attribute__((ext_vector_type(16))) _Float16 v16h;
typedef __attribute__((ext_vector_type(8)))  _Float16 v8h;
typedef __attribute__((ext_vector_type(8)))  float    v8f;

#define VOCAB 32000
#define CTX   8
#define HDIM  256
#define BATCH 512
#define NOUT  (CTX * VOCAB)   // 256000
#define BPAD  264             // LDS column stride in halves (4-bank skew)

// ---------------------------------------------------------------------------
// Kernel 1: fused  h1 = relu(sum_p E[p, x[b,p], :] + b1)
//                  h2 = relu(h1 @ W2 + b2)   stored as f16
// One block per batch row; 256 threads = one output feature each.
// ---------------------------------------------------------------------------
__global__ void __launch_bounds__(256)
mlp_front_kernel(const int* __restrict__ x, const float* __restrict__ E,
                 const float* __restrict__ b1, const float* __restrict__ W2,
                 const float* __restrict__ b2, _Float16* __restrict__ h2)
{
    __shared__ float h1[HDIM];
    const int b = blockIdx.x;
    const int d = threadIdx.x;

    float acc = b1[d];
#pragma unroll
    for (int p = 0; p < CTX; ++p) {
        const int idx = x[b * CTX + p];                 // token id, 0..31999
        acc += E[((size_t)p * VOCAB + idx) * HDIM + d]; // coalesced over d
    }
    h1[d] = fmaxf(acc, 0.0f);
    __syncthreads();

    float acc2 = b2[d];
#pragma unroll 8
    for (int k = 0; k < HDIM; ++k)
        acc2 = fmaf(h1[k], W2[k * HDIM + d], acc2);     // coalesced over d

    h2[(size_t)b * HDIM + d] = (_Float16)fmaxf(acc2, 0.0f);
}

// ---------------------------------------------------------------------------
// Kernel 2: out = h2 @ W3 + b3 via v_wmma_f32_16x16x32_f16.
// Block = 256 threads = 8 wave32. Each block covers ALL M=512 rows x 16 cols,
// so W3 streams from HBM exactly once. Wave w owns M rows [w*64, w*64+64).
//
// Fragment layouts (CDNA5 ISA 7.12.2):
//  A 16x32 f16: lane&15 = M row; lanes 0-15 hold K = {k0..k0+7, k0+16..k0+23},
//               lanes 16-31 hold K = {k0+8..k0+15, k0+24..k0+31}
//               -> two contiguous 16B loads per lane from row-major h2.
//  B 32x16 f16: lane&15 = N col; lanes 0-15 hold K = {k0..k0+7, k0+16..k0+23},
//               lanes 16-31 hold K = {k0+8..k0+15, k0+24..k0+31} (same as A)
//               -> stage W3 panel col-major in LDS (padded stride) and read
//                  two 16B pieces per lane, conflict-skewed.
//  C/D 16x16 f32: lane&15 = N col; VGPR r = M row (r, lanes<16) / (8+r).
// ---------------------------------------------------------------------------
__global__ void __launch_bounds__(256)
mlp_gemm_kernel(const _Float16* __restrict__ h2, const float* __restrict__ W3,
                const float* __restrict__ b3, float* __restrict__ out)
{
    __shared__ _Float16 Bp[16 * BPAD];   // [col][k] f16, padded: 8448 B LDS

    const int n0  = blockIdx.x * 16;
    const int tid = threadIdx.x;

    // ---- Stage W3[0:256, n0:n0+16] -> LDS f16, column-major padded --------
    {
        const int q  = tid & 3;      // which float4 within the 16-col row
        const int kr = tid >> 2;     // rows kr, kr+64, kr+128, kr+192
#pragma unroll
        for (int r = 0; r < 4; ++r) {
            const int k = kr + r * 64;
            const float4 f =
                *(const float4*)(W3 + (size_t)k * NOUT + n0 + q * 4);
            Bp[(q * 4 + 0) * BPAD + k] = (_Float16)f.x;
            Bp[(q * 4 + 1) * BPAD + k] = (_Float16)f.y;
            Bp[(q * 4 + 2) * BPAD + k] = (_Float16)f.z;
            Bp[(q * 4 + 3) * BPAD + k] = (_Float16)f.w;
        }
    }
    __syncthreads();

    const int wave   = tid >> 5;
    const int lane   = tid & 31;
    const int col    = lane & 15;    // N column within panel (B/C/D lane role)
    const int khalf  = lane >> 4;    // which K half-group this lane holds
    const int mwbase = wave * 64;    // this wave's 4 M-tiles of 16

    // Accumulators pre-loaded with bias (each D element gets b3[n] once).
    const float bias = b3[n0 + col];
    v8f acc[4];
#pragma unroll
    for (int i = 0; i < 4; ++i)
#pragma unroll
        for (int j = 0; j < 8; ++j) acc[i][j] = bias;

    // ---- K loop: 8 chunks of 32, B fragment reused across 4 M-tiles -------
#pragma unroll
    for (int k0 = 0; k0 < HDIM; k0 += 32) {
        union { v16h v; v8h h[2]; } bu;
        const _Float16* bsrc = &Bp[col * BPAD + k0 + khalf * 8];
        bu.h[0] = *(const v8h*)(bsrc);
        bu.h[1] = *(const v8h*)(bsrc + 16);
        const v16h bfrag = bu.v;

        const int kbase = k0 + khalf * 8;
#pragma unroll
        for (int i = 0; i < 4; ++i) {
            const _Float16* asrc =
                h2 + (size_t)(mwbase + i * 16 + col) * HDIM + kbase;
            union { v16h v; v8h h[2]; } au;
            au.h[0] = *(const v8h*)(asrc);        // K kbase..kbase+7
            au.h[1] = *(const v8h*)(asrc + 16);   // K kbase+16..kbase+23
            acc[i] = __builtin_amdgcn_wmma_f32_16x16x32_f16(
                false, au.v, false, bfrag, (short)0, acc[i], false, false);
        }
    }

    // ---- Epilogue: D layout -> row-major out (64B runs per half-wave) ------
#pragma unroll
    for (int i = 0; i < 4; ++i) {
        const int rowbase = mwbase + i * 16 + khalf * 8;
#pragma unroll
        for (int r = 0; r < 8; ++r)
            out[(size_t)(rowbase + r) * NOUT + n0 + col] = acc[i][r];
    }
}

// ---------------------------------------------------------------------------
extern "C" void kernel_launch(void* const* d_in, const int* in_sizes, int n_in,
                              void* d_out, int out_size, void* d_ws, size_t ws_size,
                              hipStream_t stream)
{
    (void)in_sizes; (void)n_in; (void)out_size; (void)ws_size;

    const int*   x  = (const int*)  d_in[0];  // [512, 8] token ids
    const float* E  = (const float*)d_in[1];  // [8, 32000, 256]
    const float* b1 = (const float*)d_in[2];  // [256]
    const float* W2 = (const float*)d_in[3];  // [256, 256]
    const float* b2 = (const float*)d_in[4];  // [256]
    const float* W3 = (const float*)d_in[5];  // [256, 256000]
    const float* b3 = (const float*)d_in[6];  // [256000]
    float*       out = (float*)d_out;         // [512, 8, 32000]

    _Float16* h2 = (_Float16*)d_ws;           // 512*256 f16 = 256 KB scratch

    mlp_front_kernel<<<BATCH, HDIM, 0, stream>>>(x, E, b1, W2, b2, h2);
    mlp_gemm_kernel<<<NOUT / 16, 256, 0, stream>>>(h2, W3, b3, out);
}
